// _SimpleGCN_876173328454
// MI455X (gfx1250) — matrix operand order repaired and checked by reading, not compiled, not run on hardware
//
#include <hip/hip_runtime.h>
#include <hip/hip_bf16.h>

#define NN 8192
#define INDIM 128
#define HID 32

typedef __attribute__((ext_vector_type(16))) __bf16        v16bf;
typedef __attribute__((ext_vector_type(8)))  float         v8f;
typedef __attribute__((ext_vector_type(8)))  unsigned int  v8u;

// float -> bf16, round to nearest even
__device__ __forceinline__ unsigned short f2bf(float f) {
    unsigned int u = __builtin_bit_cast(unsigned int, f);
    unsigned int r = u + 0x7FFFu + ((u >> 16) & 1u);
    return (unsigned short)(r >> 16);
}

// ---------------------------------------------------------------------------
// 1a) Fused pass over adj (one row per block):
//     - degree[i] = sum_j adj[i][j] + 1   -> dis[i] = clip(deg,1)^-0.5
//     - Abf[i][j] = bf16(adj[i][j] + (i==j))   (A_hat pre-converted)
// ---------------------------------------------------------------------------
__global__ __launch_bounds__(256) void gcn_conv_kernel(
    const float* __restrict__ adj, unsigned short* __restrict__ Abf,
    float* __restrict__ dis)
{
    __shared__ float red[256];
    const int row = blockIdx.x;
    const float* __restrict__ arow = adj + (size_t)row * NN;
    unsigned int* __restrict__ brow = (unsigned int*)(Abf + (size_t)row * NN);
    float s = 0.f;
    for (int j0 = threadIdx.x * 2; j0 < NN; j0 += 512) {
        const float2 a2 = *(const float2*)(arow + j0);
        s += a2.x + a2.y;
        float a0 = a2.x, a1 = a2.y;
        if (row == j0)     a0 += 1.0f;   // fold +I into stored A_hat
        if (row == j0 + 1) a1 += 1.0f;
        brow[j0 >> 1] = (unsigned int)f2bf(a0) | ((unsigned int)f2bf(a1) << 16);
    }
    red[threadIdx.x] = s;
    __syncthreads();
    for (int off = 128; off > 0; off >>= 1) {
        if (threadIdx.x < off) red[threadIdx.x] += red[threadIdx.x + off];
        __syncthreads();
    }
    if (threadIdx.x == 0) {
        float deg = red[0] + 1.0f;          // + self loop
        deg = deg < 1.0f ? 1.0f : deg;      // clip(deg, 1, None)
        dis[row] = rsqrtf(deg);
    }
}

// 1b) Fallback (small workspace): degree only, adj stays fp32
__global__ __launch_bounds__(256) void gcn_degree_kernel(
    const float* __restrict__ adj, float* __restrict__ dis)
{
    __shared__ float red[256];
    const int row = blockIdx.x;
    const float* arow = adj + (size_t)row * NN;
    float s = 0.f;
    for (int j = threadIdx.x; j < NN; j += 256) s += arow[j];
    red[threadIdx.x] = s;
    __syncthreads();
    for (int off = 128; off > 0; off >>= 1) {
        if (threadIdx.x < off) red[threadIdx.x] += red[threadIdx.x + off];
        __syncthreads();
    }
    if (threadIdx.x == 0) {
        float deg = red[0] + 1.0f;
        deg = deg < 1.0f ? 1.0f : deg;
        dis[row] = rsqrtf(deg);
    }
}

// ---------------------------------------------------------------------------
// 2) ZT[h][j] = bf16( dis[j] * (x[j]·W1[:,h] + b1[h]) )   (transposed store)
// ---------------------------------------------------------------------------
__global__ __launch_bounds__(256) void gcn_featA_kernel(
    const float* __restrict__ x, const float* __restrict__ W1,
    const float* __restrict__ b1, const float* __restrict__ dis,
    unsigned short* __restrict__ ZT)
{
    const int t    = blockIdx.x * 256 + threadIdx.x;
    const int hcol = t & (HID - 1);
    const int j    = t >> 5;
    const float* xr = x + (size_t)j * INDIM;
    float acc = b1[hcol];
#pragma unroll 4
    for (int d = 0; d < INDIM; ++d) acc += xr[d] * W1[d * HID + hcol];
    acc *= dis[j];
    ZT[(size_t)hcol * NN + j] = f2bf(acc);
}

// ---------------------------------------------------------------------------
// 3a) WMMA propagation, bf16 A path:
//     P[chunk][i][h] = sum_{k in chunk} Abf[i][k]*Zs[k][h]
//     One wave per (16-row M-tile, K-chunk); two v_wmma_f32_16x16x32_bf16
//     per K=32 step cover HID=32 columns. Deterministic (no atomics).
// ---------------------------------------------------------------------------
__global__ __launch_bounds__(128) void gcn_prop_bf16_kernel(
    const unsigned short* __restrict__ Abf,  // bf16 [NN][NN], +I folded
    const unsigned short* __restrict__ ZT,   // bf16 [HID][NN]
    float* __restrict__ P,                   // [nchunk][NN][HID]
    int nchunk, int kPerChunk)
{
    const int wave  = threadIdx.x >> 5;
    const int lane  = threadIdx.x & 31;
    const int wid   = blockIdx.x * 4 + wave;
    const int mtile = wid / nchunk;
    const int chunk = wid - mtile * nchunk;
    const int rbase = mtile * 16;
    const int kbeg  = chunk * kPerChunk;
    const int kend  = kbeg + kPerChunk;

    const int m  = lane & 15;     // row within tile (both lane halves)
    const int h  = lane >> 4;     // lane half selects K sub-block
    const int n0 = lane & 15;     // B/D column within 16-wide half
    const int row = rbase + m;

    // packed bf16 pairs of row `row`
    const unsigned int* __restrict__ arow2 =
        (const unsigned int*)(Abf + (size_t)row * NN);
    const unsigned short* __restrict__ zt0 = ZT + (size_t)n0 * NN;
    const unsigned short* __restrict__ zt1 = ZT + (size_t)(n0 + 16) * NN;

    v8f c0 = {}; v8f c1 = {};

    for (int kk = kbeg; kk < kend; kk += 32) {
        __builtin_prefetch(arow2 + (kk >> 1) + 128, 0, 1);  // 512B ahead

        // A fragment: VGPR v holds packed pair at K = kb(v)
        //   kb(v) = kk + (v&4?16:0) + h*8 + (v&3)*2  -> pair index kb/2
        v8u ap;
#pragma unroll
        for (int v = 0; v < 8; ++v)
            ap[v] = arow2[(kk >> 1) + ((v & 4) << 1) + h * 4 + (v & 3)];
        const v16bf af = __builtin_bit_cast(v16bf, ap);

        // B fragments: VGPR v, lane half h -> K = kk + (v&4?16:0) + h*8 + (v&3)*2
        v8u bp0, bp1;
#pragma unroll
        for (int v = 0; v < 8; ++v) {
            const int k = kk + ((v & 4) << 2) + h * 8 + (v & 3) * 2;
            bp0[v] = *(const unsigned int*)(zt0 + k);
            bp1[v] = *(const unsigned int*)(zt1 + k);
        }
        const v16bf bf0 = __builtin_bit_cast(v16bf, bp0);
        const v16bf bf1 = __builtin_bit_cast(v16bf, bp1);

        c0 = __builtin_amdgcn_wmma_f32_16x16x32_bf16(
                 false, af, false, bf0, (short)0, c0, false, false);
        c1 = __builtin_amdgcn_wmma_f32_16x16x32_bf16(
                 false, af, false, bf1, (short)0, c1, false, false);
    }

    // D layout: VGPR r -> M = r + 8*half, N = lane&15 (+16 for c1)
    float* __restrict__ prow = P + ((size_t)chunk * NN + rbase) * HID;
#pragma unroll
    for (int r = 0; r < 8; ++r) {
        const int mrow = r + 8 * h;
        prow[(size_t)mrow * HID + n0]      = c0[r];
        prow[(size_t)mrow * HID + 16 + n0] = c1[r];
    }
}

// ---------------------------------------------------------------------------
// 3b) WMMA propagation, fp32-A fallback (converts on the fly, +I on the fly)
// ---------------------------------------------------------------------------
__global__ __launch_bounds__(128) void gcn_prop_f32_kernel(
    const float* __restrict__ adj,
    const unsigned short* __restrict__ ZT,
    float* __restrict__ P,
    int nchunk, int kPerChunk)
{
    const int wave  = threadIdx.x >> 5;
    const int lane  = threadIdx.x & 31;
    const int wid   = blockIdx.x * 4 + wave;
    const int mtile = wid / nchunk;
    const int chunk = wid - mtile * nchunk;
    const int rbase = mtile * 16;
    const int kbeg  = chunk * kPerChunk;
    const int kend  = kbeg + kPerChunk;

    const int m  = lane & 15;
    const int h  = lane >> 4;
    const int n0 = lane & 15;
    const int row = rbase + m;

    const float* __restrict__ arow = adj + (size_t)row * NN;
    const unsigned short* __restrict__ zt0 = ZT + (size_t)n0 * NN;
    const unsigned short* __restrict__ zt1 = ZT + (size_t)(n0 + 16) * NN;

    v8f c0 = {}; v8f c1 = {};

    for (int kk = kbeg; kk < kend; kk += 32) {
        __builtin_prefetch(arow + kk + 256, 0, 1);

        v8u ap;
#pragma unroll
        for (int v = 0; v < 8; ++v) {
            const int kb = kk + ((v & 4) << 2) + h * 8 + (v & 3) * 2;
            const float2 a2 = *(const float2*)(arow + kb);
            float a0 = a2.x, a1 = a2.y;
            if (row == kb)     a0 += 1.0f;
            if (row == kb + 1) a1 += 1.0f;
            ap[v] = (unsigned int)f2bf(a0) | ((unsigned int)f2bf(a1) << 16);
        }
        const v16bf af = __builtin_bit_cast(v16bf, ap);

        v8u bp0, bp1;
#pragma unroll
        for (int v = 0; v < 8; ++v) {
            const int k = kk + ((v & 4) << 2) + h * 8 + (v & 3) * 2;
            bp0[v] = *(const unsigned int*)(zt0 + k);
            bp1[v] = *(const unsigned int*)(zt1 + k);
        }
        const v16bf bf0 = __builtin_bit_cast(v16bf, bp0);
        const v16bf bf1 = __builtin_bit_cast(v16bf, bp1);

        c0 = __builtin_amdgcn_wmma_f32_16x16x32_bf16(
                 false, af, false, bf0, (short)0, c0, false, false);
        c1 = __builtin_amdgcn_wmma_f32_16x16x32_bf16(
                 false, af, false, bf1, (short)0, c1, false, false);
    }

    float* __restrict__ prow = P + ((size_t)chunk * NN + rbase) * HID;
#pragma unroll
    for (int r = 0; r < 8; ++r) {
        const int mrow = r + 8 * h;
        prow[(size_t)mrow * HID + n0]      = c0[r];
        prow[(size_t)mrow * HID + 16 + n0] = c1[r];
    }
}

// ---------------------------------------------------------------------------
// 4) h1 = relu(dis[j] * sum_c P[c][j][:]);  ZT[h][j] = bf16(dis[j]*(h1·W2+b2))
// ---------------------------------------------------------------------------
__global__ __launch_bounds__(256) void gcn_featB_kernel(
    const float* __restrict__ P, int nchunk,
    const float* __restrict__ W2, const float* __restrict__ b2,
    const float* __restrict__ dis, unsigned short* __restrict__ ZT)
{
    __shared__ float hf[8][HID];
    const int hcol = threadIdx.x & (HID - 1);
    const int jl   = threadIdx.x >> 5;
    const int j    = blockIdx.x * 8 + jl;
    const float dj = dis[j];

    float s = 0.f;
    for (int c = 0; c < nchunk; ++c)
        s += P[((size_t)c * NN + j) * HID + hcol];
    s *= dj;
    hf[jl][hcol] = s > 0.f ? s : 0.f;
    __syncthreads();

    float acc = b2[hcol];
#pragma unroll
    for (int g = 0; g < HID; ++g) acc += hf[jl][g] * W2[g * HID + hcol];
    acc *= dj;
    ZT[(size_t)hcol * NN + j] = f2bf(acc);
}

// ---------------------------------------------------------------------------
// 5) logits[i] = relu(dis[i] * sum_c P[c][i][:]) · W3 + b3
// ---------------------------------------------------------------------------
__global__ __launch_bounds__(256) void gcn_final_kernel(
    const float* __restrict__ P, int nchunk,
    const float* __restrict__ W3, const float* __restrict__ b3,
    const float* __restrict__ dis, float* __restrict__ out)
{
    const int i = blockIdx.x * 256 + threadIdx.x;
    const float di = dis[i];
    float acc = b3[0];
#pragma unroll
    for (int g = 0; g < HID; ++g) {
        float s = 0.f;
        for (int c = 0; c < nchunk; ++c)
            s += P[((size_t)c * NN + i) * HID + g];
        s *= di;
        acc += (s > 0.f ? s : 0.f) * W3[g];
    }
    out[i] = acc;
}

// ---------------------------------------------------------------------------
extern "C" void kernel_launch(void* const* d_in, const int* in_sizes, int n_in,
                              void* d_out, int out_size, void* d_ws, size_t ws_size,
                              hipStream_t stream) {
    const float* x   = (const float*)d_in[0];
    const float* adj = (const float*)d_in[1];
    const float* W1  = (const float*)d_in[2];
    const float* b1  = (const float*)d_in[3];
    const float* W2  = (const float*)d_in[4];
    const float* b2  = (const float*)d_in[5];
    const float* W3  = (const float*)d_in[6];
    const float* b3  = (const float*)d_in[7];
    float* out = (float*)d_out;

    const size_t per  = (size_t)NN * HID * sizeof(float);            // 1 MB / slice
    const size_t head = 32 * 1024 + 512 * 1024;                      // dis + ZT
    const size_t abfBytes = (size_t)NN * NN * sizeof(unsigned short); // 128 MB

    char* ws = (char*)d_ws;
    const bool bf16Path = ws_size >= abfBytes + head + per;

    unsigned short* Abf = (unsigned short*)ws;                        // bf16 path only
    char* base = bf16Path ? (ws + abfBytes) : ws;
    float*          dis = (float*)base;                               // 32 KB
    unsigned short* ZT  = (unsigned short*)(base + 32 * 1024);        // 512 KB
    float*          P   = (float*)(base + head);

    // K-split factor: as many deterministic partial slices as workspace allows
    const size_t used  = (bf16Path ? abfBytes : 0) + head;
    const size_t avail = ws_size > used ? ws_size - used : 0;
    int nchunk = 1;
    if      (avail >= 8 * per) nchunk = 8;
    else if (avail >= 4 * per) nchunk = 4;
    else if (avail >= 2 * per) nchunk = 2;
    const int kPerChunk = NN / nchunk;
    const int propBlocks = (NN / 16) * nchunk / 4;   // 4 waves / block

    if (bf16Path) {
        gcn_conv_kernel<<<NN, 256, 0, stream>>>(adj, Abf, dis);
        gcn_featA_kernel<<<NN * HID / 256, 256, 0, stream>>>(x, W1, b1, dis, ZT);
        gcn_prop_bf16_kernel<<<propBlocks, 128, 0, stream>>>(Abf, ZT, P, nchunk, kPerChunk);
        gcn_featB_kernel<<<NN / 8, 256, 0, stream>>>(P, nchunk, W2, b2, dis, ZT);
        gcn_prop_bf16_kernel<<<propBlocks, 128, 0, stream>>>(Abf, ZT, P, nchunk, kPerChunk);
        gcn_final_kernel<<<NN / 256, 256, 0, stream>>>(P, nchunk, W3, b3, dis, out);
    } else {
        gcn_degree_kernel<<<NN, 256, 0, stream>>>(adj, dis);
        gcn_featA_kernel<<<NN * HID / 256, 256, 0, stream>>>(x, W1, b1, dis, ZT);
        gcn_prop_f32_kernel<<<propBlocks, 128, 0, stream>>>(adj, ZT, P, nchunk, kPerChunk);
        gcn_featB_kernel<<<NN / 8, 256, 0, stream>>>(P, nchunk, W2, b2, dis, ZT);
        gcn_prop_f32_kernel<<<propBlocks, 128, 0, stream>>>(adj, ZT, P, nchunk, kPerChunk);
        gcn_final_kernel<<<NN / 256, 256, 0, stream>>>(P, nchunk, W3, b3, dis, out);
    }
}
